// GH200OptimizedAttention_40175124087064
// MI455X (gfx1250) — matrix operand order repaired and checked by reading, not compiled, not run on hardware
//
#include <hip/hip_runtime.h>

// ---------------------------------------------------------------------------
// MI455X (gfx1250) attention: bf16 WMMA GEMMs (TDM-staged, double-buffered LDS)
// + fp32 softmax.  B=4, S=2048, D=1024, U=1024.
// d_out = [output fp32 (B*S*D)] ++ [weights fp32 (B*S*S)]
// All GEMMs use B stored [N][K] row-major ("transposed-B"), so LDS staging is
// pure 2-D row copies -> Tensor Data Mover friendly.
// ---------------------------------------------------------------------------

typedef __attribute__((ext_vector_type(16))) __bf16       v16bf;
typedef __attribute__((ext_vector_type(8)))  float        v8f;
typedef __attribute__((ext_vector_type(4)))  unsigned int u32x4;
typedef __attribute__((ext_vector_type(8)))  int          i32x8;
typedef __attribute__((ext_vector_type(4)))  int          i32x4;

#if __has_builtin(__builtin_amdgcn_tensor_load_to_lds)
#define TDM_AVAILABLE 1
#else
#define TDM_AVAILABLE 0
#endif

static constexpr int Bb = 4;
static constexpr int Ss = 2048;
static constexpr int Dd = 1024;
static constexpr int Uu = 1024;
static constexpr long BS = (long)Bb * Ss;          // 8192
static constexpr long OUT_ELEMS = BS * Dd;         // fp32 output elems

__device__ __forceinline__ unsigned short bf16_bits(float f) {
    union { float f; unsigned u; } x; x.f = f;
    unsigned r = x.u + 0x7FFFu + ((x.u >> 16) & 1u);   // round-to-nearest-even
    return (unsigned short)(r >> 16);
}
__device__ __forceinline__ __bf16 to_bf16(float f) {
    unsigned short h = bf16_bits(f);
    union { unsigned short s; __bf16 b; } y; y.s = h;
    return y.b;
}

// ---------------------------------------------------------------------------
// fp32 -> bf16 (straight)
// ---------------------------------------------------------------------------
__global__ __launch_bounds__(256)
void cvt_f32_bf16(const float* __restrict__ in, __bf16* __restrict__ out, long n) {
    long i = (((long)blockIdx.x * blockDim.x) + threadIdx.x) * 2;
    if (i + 1 < n) {
        unsigned lo = bf16_bits(in[i]);
        unsigned hi = bf16_bits(in[i + 1]);
        *(unsigned*)(&out[i]) = lo | (hi << 16);
    }
}

// ---------------------------------------------------------------------------
// fp32 [R][Cc] -> bf16 transposed [Cc][R], 32x32 LDS tiles
// ---------------------------------------------------------------------------
__global__ __launch_bounds__(256)
void cvt_f32_bf16_T(const float* __restrict__ in, __bf16* __restrict__ out,
                    int R, int Cc) {
    __shared__ float tile[32][33];
    const int bx = blockIdx.x * 32;   // column tile of input
    const int by = blockIdx.y * 32;   // row tile of input
    const int tx = threadIdx.x & 31;
    const int ty = threadIdx.x >> 5;  // 0..7
    #pragma unroll
    for (int r = 0; r < 4; ++r)
        tile[ty + 8 * r][tx] = in[(size_t)(by + ty + 8 * r) * Cc + bx + tx];
    __syncthreads();
    #pragma unroll
    for (int r = 0; r < 4; ++r)
        out[(size_t)(bx + ty + 8 * r) * R + by + tx] = to_bf16(tile[tx][ty + 8 * r]);
}

// ---------------------------------------------------------------------------
// TDM: async-load a 128x32 bf16 tile (row stride = strideElems) into LDS with
// 16B padding per 64B row => LDS row pitch 80B (matches BKP=40 halves).
// ---------------------------------------------------------------------------
#if TDM_AVAILABLE
__device__ __forceinline__ void tdm_load_tile_128x32(unsigned long long gaddr,
                                                     unsigned lds_byte_addr,
                                                     int strideElems) {
    u32x4 g0;
    g0.x = 1u;                                                  // count=1, user D#
    g0.y = lds_byte_addr;                                       // LDS dest (bytes)
    g0.z = (unsigned)(gaddr & 0xFFFFFFFFull);                   // global_addr lo
    g0.w = (unsigned)((gaddr >> 32) & 0x1FFFFFFull) | (2u << 30); // hi | type=2
    i32x8 g1;
    g1[0] = (int)((1u << 16)     // data_size = 2 bytes
                | (1u << 20)     // pad_enable
                | (3u << 22)     // pad_interval: every 16 DWORDs (64B)
                | (3u << 25));   // pad_amount: 4 DWORDs (16B)
    g1[1] = (int)(32u << 16);    // tensor_dim0 = 32
    g1[2] = (int)(128u << 16);   // tensor_dim1 = 128
    g1[3] = (int)(32u << 16);    // tile_dim0 = 32
    g1[4] = 128;                 // tile_dim1 = 128 (tile_dim2 = 0)
    g1[5] = strideElems;         // tensor_dim0_stride
    g1[6] = 0;
    g1[7] = 0;
    i32x4 z4 = {0, 0, 0, 0};
    i32x8 z8 = {0, 0, 0, 0, 0, 0, 0, 0};
    // 6-arg toolchain form: (g0, g1, g2, g3, extra, cpol); 2-D tile => groups
    // 2/3 disabled (zero).
    __builtin_amdgcn_tensor_load_to_lds(g0, g1, z4, z4, z8, 0);
}
#endif

// ---------------------------------------------------------------------------
// Tiled GEMM: C[M,N] = alpha * A[M,K](bf16) x Bst[N,K](bf16)^T (+ bias)
// Tile: 128x128, BK=32. 256 threads = 8 waves (2 M x 4 N), each wave owns a
// 4x2 grid of 16x16 v_wmma_f32_16x16x32_bf16 accumulators.
// OUTTRANS: write C transposed as [N][M] (ldc = stride between N rows),
//           packed 8xbf16 b128 stores.
// ---------------------------------------------------------------------------
template<bool OUTBF16, bool BIAS, bool OUTTRANS>
__global__ __launch_bounds__(256)
void gemm_bf16_wmma(const __bf16* __restrict__ A, const __bf16* __restrict__ Bm,
                    void* __restrict__ C,
                    const float* __restrict__ bias,
                    const float* __restrict__ alphaPtr,
                    int K, int lda, int ldb, int ldc)
{
    constexpr int BKP = 40;                 // 32 + 8 halves pad -> 80B row pitch
    __shared__ __bf16 lA[2][128 * BKP];
    __shared__ __bf16 lB[2][128 * BKP];

    const int t    = threadIdx.x;
    const int lane = t & 31;
    const int wave = t >> 5;
    const int wm   = wave & 1;
    const int wn   = wave >> 1;
    const int lr   = lane & 15;
    const int kg   = lane >> 4;

    const int bm0 = blockIdx.y * 128;
    const int bn0 = blockIdx.x * 128;

    const v8f vzero = {0.f,0.f,0.f,0.f,0.f,0.f,0.f,0.f};
    v8f acc[4][2];
    #pragma unroll
    for (int i = 0; i < 4; ++i)
        #pragma unroll
        for (int j = 0; j < 2; ++j) acc[i][j] = vzero;

#if TDM_AVAILABLE
    const unsigned long long gA = (unsigned long long)(const void*)(A + (size_t)bm0 * lda);
    const unsigned long long gB = (unsigned long long)(const void*)(Bm + (size_t)bn0 * ldb);
    const unsigned ldsA0 = (unsigned)(size_t)(void*)&lA[0][0];
    const unsigned ldsA1 = (unsigned)(size_t)(void*)&lA[1][0];
    const unsigned ldsB0 = (unsigned)(size_t)(void*)&lB[0][0];
    const unsigned ldsB1 = (unsigned)(size_t)(void*)&lB[1][0];
    if (t < 32) {                               // wave 0 drives the TDM
        tdm_load_tile_128x32(gA, ldsA0, lda);
        tdm_load_tile_128x32(gB, ldsB0, ldb);
    }
#else
    const int srow = t >> 1;
    const int skc  = (t & 1) * 16;
#endif

    int buf = 0;
    for (int k0 = 0; k0 < K; k0 += 32, buf ^= 1) {
#if TDM_AVAILABLE
        if (t < 32) {
            if (k0 + 32 < K) {
                const unsigned long long koff = (unsigned long long)(k0 + 32) * 2ull;
                tdm_load_tile_128x32(gA + koff, buf ? ldsA0 : ldsA1, lda);
                tdm_load_tile_128x32(gB + koff, buf ? ldsB0 : ldsB1, ldb);
                __builtin_amdgcn_s_wait_tensorcnt(2);  // tile k complete; k+1 in flight
            } else {
                __builtin_amdgcn_s_wait_tensorcnt(0);
            }
        }
        __syncthreads();
#else
        {
            const u32x4* pA = (const u32x4*)(A + (size_t)(bm0 + srow) * lda + k0 + skc);
            u32x4 a0 = pA[0], a1 = pA[1];
            *(u32x4*)(&lA[buf][srow * BKP + skc])     = a0;
            *(u32x4*)(&lA[buf][srow * BKP + skc + 8]) = a1;
            const u32x4* pB = (const u32x4*)(Bm + (size_t)(bn0 + srow) * ldb + k0 + skc);
            u32x4 b0 = pB[0], b1 = pB[1];
            *(u32x4*)(&lB[buf][srow * BKP + skc])     = b0;
            *(u32x4*)(&lB[buf][srow * BKP + skc + 8]) = b1;
        }
        __syncthreads();
#endif

        union Frag { u32x4 u[2]; v16bf v; };
        Frag af[4], bfr[2];
        #pragma unroll
        for (int i = 0; i < 4; ++i) {
            // A 16x32 bf16: lane row=lr, K chunks {kg*8..+7} and {16+kg*8..+7}
            const __bf16* p = &lA[buf][(wm * 64 + i * 16 + lr) * BKP + kg * 8];
            af[i].u[0] = *(const u32x4*)(p);
            af[i].u[1] = *(const u32x4*)(p + 16);
        }
        #pragma unroll
        for (int j = 0; j < 2; ++j) {
            // B 32x16 bf16: lane col=lr, K chunks {kg*8..+7} and {16+kg*8..+7}
            const __bf16* p = &lB[buf][(wn * 32 + j * 16 + lr) * BKP + kg * 8];
            bfr[j].u[0] = *(const u32x4*)(p);
            bfr[j].u[1] = *(const u32x4*)(p + 16);
        }

        #pragma unroll
        for (int i = 0; i < 4; ++i)
            #pragma unroll
            for (int j = 0; j < 2; ++j)
                acc[i][j] = __builtin_amdgcn_wmma_f32_16x16x32_bf16(
                    false, af[i].v, false, bfr[j].v,
                    (short)0, acc[i][j], false, false);

        __syncthreads();
    }

    const float alpha = alphaPtr ? *alphaPtr : 1.0f;

    // C/D 16x16 f32 layout: VGPR r -> row (kg*8 + r), col = lr
    #pragma unroll
    for (int i = 0; i < 4; ++i) {
        #pragma unroll
        for (int j = 0; j < 2; ++j) {
            const int col = bn0 + wn * 32 + j * 16 + lr;
            const int rowbase = bm0 + wm * 64 + i * 16 + kg * 8;
            if (OUTTRANS) {
                // packed transposed store: Ct[col][rowbase..rowbase+7], 16B
                u32x4 pk;
                #pragma unroll
                for (int r = 0; r < 4; ++r) {
                    unsigned lo = bf16_bits(acc[i][j][2 * r]     * alpha);
                    unsigned hi = bf16_bits(acc[i][j][2 * r + 1] * alpha);
                    pk[r] = lo | (hi << 16);
                }
                *(u32x4*)((__bf16*)C + (size_t)col * ldc + rowbase) = pk;
            } else {
                const float badd = BIAS ? bias[col] : 0.0f;
                #pragma unroll
                for (int r = 0; r < 8; ++r) {
                    const float v = acc[i][j][r] * alpha + badd;
                    if (OUTBF16) ((__bf16*)C)[(size_t)(rowbase + r) * ldc + col] = to_bf16(v);
                    else         ((float*)C)[(size_t)(rowbase + r) * ldc + col]  = v;
                }
            }
        }
    }
}

// ---------------------------------------------------------------------------
// Row softmax over S elems: in-place fp32 + bf16 copy for the next GEMM.
// ---------------------------------------------------------------------------
__global__ __launch_bounds__(256)
void softmax_rows(float* __restrict__ scores, __bf16* __restrict__ wbf, int S)
{
    const long row = blockIdx.x;
    float* p = scores + row * (long)S;
    __bf16* q = wbf + row * (long)S;
    __shared__ float red[256];
    const int t = threadIdx.x;

    float m = -3.402823466e+38f;
    for (int i = t; i < S; i += 256) m = fmaxf(m, p[i]);
    red[t] = m; __syncthreads();
    for (int off = 128; off > 0; off >>= 1) {
        if (t < off) red[t] = fmaxf(red[t], red[t + off]);
        __syncthreads();
    }
    m = red[0]; __syncthreads();

    float s = 0.0f;
    for (int i = t; i < S; i += 256) {
        float e = __expf(p[i] - m);
        p[i] = e;
        s += e;
    }
    red[t] = s; __syncthreads();
    for (int off = 128; off > 0; off >>= 1) {
        if (t < off) red[t] += red[t + off];
        __syncthreads();
    }
    const float inv = 1.0f / red[0];

    for (int i = t; i < S; i += 256) {
        float w = p[i] * inv;
        p[i] = w;
        q[i] = to_bf16(w);
    }
}

// ---------------------------------------------------------------------------
extern "C" void kernel_launch(void* const* d_in, const int* in_sizes, int n_in,
                              void* d_out, int out_size, void* d_ws, size_t ws_size,
                              hipStream_t stream)
{
    (void)in_sizes; (void)n_in; (void)out_size; (void)ws_size;

    const float* inputs = (const float*)d_in[0];   // [B,S,D]
    const float* Wq     = (const float*)d_in[1];   // [D,U]
    const float* Wk     = (const float*)d_in[2];
    const float* Wv     = (const float*)d_in[3];
    const float* Wo     = (const float*)d_in[4];   // [U,D]
    const float* bo     = (const float*)d_in[5];   // [D]
    const float* scale  = (const float*)d_in[6];   // [1]

    float* out_f   = (float*)d_out;                // [B,S,D]
    float* weights = out_f + OUT_ELEMS;            // [B,S,S]

    char* ws = (char*)d_ws;
    size_t off = 0;
    auto carve = [&](size_t bytes) { void* p = ws + off; off += (bytes + 255) & ~(size_t)255; return p; };
    __bf16* Xb   = (__bf16*)carve(BS * Dd * 2);              // 16 MB
    __bf16* WqT  = (__bf16*)carve((size_t)Dd * Uu * 2);      // [U][D]
    __bf16* WkT  = (__bf16*)carve((size_t)Dd * Uu * 2);
    __bf16* WvT  = (__bf16*)carve((size_t)Dd * Uu * 2);
    __bf16* WoT  = (__bf16*)carve((size_t)Uu * Dd * 2);      // [D][U]
    __bf16* Qb   = (__bf16*)carve(BS * Uu * 2);              // [BS][U]
    __bf16* Kb   = (__bf16*)carve(BS * Uu * 2);              // [BS][U]
    __bf16* VbT  = (__bf16*)carve(BS * Uu * 2);              // [U][BS]
    __bf16* Wbf  = (__bf16*)carve((size_t)Bb * Ss * Ss * 2); // 32 MB
    __bf16* Ctxb = (__bf16*)carve(BS * Uu * 2);              // [BS][U]

    // ---- staging conversions ----
    {
        long n = BS * Dd;
        cvt_f32_bf16<<<dim3((unsigned)((n / 2 + 255) / 256)), 256, 0, stream>>>(inputs, Xb, n);
        dim3 gT(Uu / 32, Dd / 32);
        cvt_f32_bf16_T<<<gT, 256, 0, stream>>>(Wq, WqT, Dd, Uu);   // -> [U][D]
        cvt_f32_bf16_T<<<gT, 256, 0, stream>>>(Wk, WkT, Dd, Uu);
        cvt_f32_bf16_T<<<gT, 256, 0, stream>>>(Wv, WvT, Dd, Uu);
        dim3 gTo(Dd / 32, Uu / 32);
        cvt_f32_bf16_T<<<gTo, 256, 0, stream>>>(Wo, WoT, Uu, Dd);  // -> [D][U]
    }

    // ---- QKV projections: [8192,1024] x W^T -> bf16 (V written transposed) ----
    {
        dim3 grid(Uu / 128, (unsigned)(BS / 128));
        gemm_bf16_wmma<true, false, false><<<grid, 256, 0, stream>>>(
            Xb, WqT, Qb, nullptr, nullptr, Dd, Dd, Dd, Uu);
        gemm_bf16_wmma<true, false, false><<<grid, 256, 0, stream>>>(
            Xb, WkT, Kb, nullptr, nullptr, Dd, Dd, Dd, Uu);
        gemm_bf16_wmma<true, false, true><<<grid, 256, 0, stream>>>(
            Xb, WvT, VbT, nullptr, nullptr, Dd, Dd, Dd, (int)BS);  // -> [U][BS]
    }

    // ---- scores = scale * Q x K^T -> fp32 into d_out weights region ----
    for (int b = 0; b < Bb; ++b) {
        dim3 grid(Ss / 128, Ss / 128);
        gemm_bf16_wmma<false, false, false><<<grid, 256, 0, stream>>>(
            Qb + (size_t)b * Ss * Uu, Kb + (size_t)b * Ss * Uu,
            weights + (size_t)b * Ss * Ss,
            nullptr, scale, Uu, Uu, Uu, Ss);
    }

    // ---- softmax rows (in-place fp32 + bf16 copy) ----
    softmax_rows<<<dim3((unsigned)(Bb * Ss)), 256, 0, stream>>>(weights, Wbf, Ss);

    // ---- context = weights x V  (B = V^T stored [U][BS]) -> bf16 ----
    for (int b = 0; b < Bb; ++b) {
        dim3 grid(Uu / 128, Ss / 128);
        gemm_bf16_wmma<true, false, false><<<grid, 256, 0, stream>>>(
            Wbf + (size_t)b * Ss * Ss, VbT + (size_t)b * Ss,
            Ctxb + (size_t)b * Ss * Uu,
            nullptr, nullptr, Ss, Ss, (int)BS, Uu);
    }

    // ---- output = context x Wo + bo  (B = Wo^T stored [D][U]) -> fp32 ----
    {
        dim3 grid(Dd / 128, (unsigned)(BS / 128));
        gemm_bf16_wmma<false, true, false><<<grid, 256, 0, stream>>>(
            Ctxb, WoT, out_f, bo, nullptr, Uu, Uu, Uu, Dd);
    }
}
